// RBFKANLayer_3968549781593
// MI455X (gfx1250) — matrix operand order repaired and checked by reading, not compiled, not run on hardware
//
#include <hip/hip_runtime.h>
#include <hip/hip_bf16.h>

// CDNA5 WMMA / TDM vector types
typedef __attribute__((ext_vector_type(16))) __bf16       v16bf;
typedef __attribute__((ext_vector_type(8)))  __bf16       v8bf;
typedef __attribute__((ext_vector_type(8)))  float        v8f;
typedef __attribute__((ext_vector_type(4)))  unsigned int u32x4;
typedef __attribute__((ext_vector_type(8)))  int          i32x8;
typedef __attribute__((ext_vector_type(4)))  int          i32x4;

static constexpr int   IN_DIM  = 256;    // K of GEMM1
static constexpr int   C_DIM   = 1024;   // centers
static constexpr int   OUT_DIM = 512;    // output cols
static constexpr int   BM      = 128;    // rows per workgroup (8 waves x 16)
static constexpr int   BN      = 128;    // out cols per workgroup
static constexpr int   BC      = 32;     // centers per chunk == K of one stage-2 WMMA
static constexpr int   NWAVES  = 8;
static constexpr int   NCHUNK  = C_DIM / BC;  // 32
static constexpr float ALPHA_F = 1.0f;

// Padded LDS row strides (in bf16 elements): +8 elements (16B) per row kills
// the 16-way (512B stride) / 4-way (64B stride) bank conflicts on ds_load_b128.
static constexpr int CL_STRIDE = IN_DIM + 8;  // 264 -> lane bank ofs = 4*ln
static constexpr int WL_STRIDE = BC + 8;      // 40  -> lane bank ofs = 20*ln mod 64 (distinct)
static constexpr int BS_STRIDE = BC + 8;      // 40

__device__ __forceinline__ __bf16 tobf(float f) { return (__bf16)f; }

// two 16B halves 16 elements apart (table k order) -> one WMMA operand
__device__ __forceinline__ v16bf ld_frag(const __bf16* p) {
    v8bf lo = *(const v8bf*)p;
    v8bf hi = *(const v8bf*)(p + 16);
    v16bf r;
    #pragma unroll
    for (int i = 0; i < 8; ++i) { r[i] = lo[i]; r[8 + i] = hi[i]; }
    return r;
}

// ------------------------------------------------------------------
// TDM: 2D bf16 tile load global -> LDS with LDS row padding.
// D# per CDNA5 ISA ch.8. padw = pad_enable|interval|amount bits for g1[0].
// ------------------------------------------------------------------
__device__ __forceinline__ void tdm_load_bf16_2d(unsigned lds_off, const void* gptr,
                                                 unsigned tensor_d0, unsigned tensor_d1,
                                                 unsigned tile_d0,   unsigned tile_d1,
                                                 unsigned long long stride0_elems,
                                                 unsigned padw)
{
    const unsigned long long ga = (unsigned long long)gptr;
    u32x4 g0;
    g0[0] = 1u;                                              // count=1, user descriptor
    g0[1] = lds_off;                                         // LDS byte address
    g0[2] = (unsigned)(ga & 0xFFFFFFFFu);                    // global_addr[31:0]
    g0[3] = (unsigned)((ga >> 32) & 0x1FFFFFFu) | (2u << 30);// global_addr[56:32] | type=2
    i32x8 g1;
    g1[0] = (int)((1u << 16) | padw);                        // data_size=2B + pad ctl
    g1[1] = (int)((tensor_d0 & 0xFFFFu) << 16);
    g1[2] = (int)(((tensor_d0 >> 16) & 0xFFFFu) | ((tensor_d1 & 0xFFFFu) << 16));
    g1[3] = (int)(((tensor_d1 >> 16) & 0xFFFFu) | ((tile_d0 & 0xFFFFu) << 16));
    g1[4] = (int)(tile_d1 & 0xFFFFu);
    g1[5] = (int)(stride0_elems & 0xFFFFFFFFull);
    g1[6] = (int)((stride0_elems >> 32) & 0xFFFFull);
    g1[7] = 0;
    i32x4 z4 = {0, 0, 0, 0};
#if __clang_major__ >= 23
    i32x8 z8 = {0, 0, 0, 0, 0, 0, 0, 0};
    __builtin_amdgcn_tensor_load_to_lds(g0, g1, z4, z4, z8, 0);
#else
    __builtin_amdgcn_tensor_load_to_lds(g0, g1, z4, z4, 0);
#endif
}

// pad 4 DWORDs (16B) after every 128 DWORDs (one 256-elem centers row)
static constexpr unsigned PAD_C = (1u << 20) | (6u << 22) | (3u << 25);
// pad 4 DWORDs (16B) after every 16 DWORDs (one 32-elem weights row)
static constexpr unsigned PAD_W = (1u << 20) | (3u << 22) | (3u << 25);

// ------------------------------------------------------------------
// Prep 1: bf16 copies.  Cb[c][k]: [1024][256].  Wt[j][c]: [512][1024]
// (transposed so each (outBase,cc) chunk is a 128x32 tile, stride 1024).
// ------------------------------------------------------------------
__global__ __launch_bounds__(256, 1)
void prep_bf16(const float* __restrict__ centers, const float* __restrict__ weights,
               __bf16* __restrict__ Cb, __bf16* __restrict__ Wt)
{
    const int idx = blockIdx.x * 256 + threadIdx.x;
    if (idx < C_DIM * IN_DIM) {
        Cb[idx] = tobf(centers[idx]);
    } else {
        const int o = idx - C_DIM * IN_DIM;
        const int j = o >> 10;
        const int c = o & (C_DIM - 1);
        Wt[o] = tobf(weights[(size_t)c * OUT_DIM + j]);
    }
}

// ------------------------------------------------------------------
// Prep 2: exact-f32 squared norms, once globally (not per block).
// ------------------------------------------------------------------
__global__ __launch_bounds__(256, 1)
void prep_norms(const float* __restrict__ x, const float* __restrict__ centers,
                float* __restrict__ xsq, float* __restrict__ csq, int n)
{
    const int i = blockIdx.x * 256 + threadIdx.x;
    const float* row;
    float* dst;
    if (i < n)                  { row = x + (size_t)i * IN_DIM;            dst = xsq + i;     }
    else if (i < n + C_DIM)     { row = centers + (size_t)(i - n) * IN_DIM; dst = csq + i - n; }
    else return;
    const float4* r4 = (const float4*)row;
    float s = 0.f;
    #pragma unroll 8
    for (int k = 0; k < IN_DIM / 4; ++k) {
        float4 v = r4[k];
        s += v.x * v.x + v.y * v.y + v.z * v.z + v.w * v.w;
    }
    *dst = s;
}

// ------------------------------------------------------------------
// Main fused kernel: TDM double-buffered staging + WMMA bf16 chains
// ------------------------------------------------------------------
__global__ __launch_bounds__(256, 1)
void rbf_kan_fused_tdm(const float* __restrict__ x,
                       const __bf16* __restrict__ Cb,
                       const __bf16* __restrict__ Wt,
                       const float* __restrict__ csq_g,
                       const float* __restrict__ xsq_g,
                       float* __restrict__ out)
{
    __shared__ __align__(16) __bf16 Clds[2][BC * CL_STRIDE];    // 2 x 16.5 KB
    __shared__ __align__(16) __bf16 Wlds[2][BN * WL_STRIDE];    // 2 x 10   KB
    __shared__ __align__(16) __bf16 Bscr[NWAVES][16 * BS_STRIDE]; //   10   KB

    const int tid      = threadIdx.x;
    const int wave     = tid >> 5;
    const int lane     = tid & 31;
    const int hi       = lane >> 4;
    const int ln       = lane & 15;
    const int rowBlock = blockIdx.x * BM;
    const int outBase  = blockIdx.y * BN;

    // kick off TDM for chunk 0 immediately
    if (wave == 0) {
        const unsigned cl0 = (unsigned)(uintptr_t)(void*)&Clds[0][0];
        const unsigned wl0 = (unsigned)(uintptr_t)(void*)&Wlds[0][0];
        tdm_load_bf16_2d(cl0, Cb, (unsigned)(C_DIM * IN_DIM), 1u,
                         (unsigned)(BC * IN_DIM), 1u,
                         (unsigned long long)(BC * IN_DIM), PAD_C);
        tdm_load_bf16_2d(wl0, Wt + (size_t)outBase * C_DIM, (unsigned)C_DIM,
                         (unsigned)OUT_DIM, (unsigned)BC, (unsigned)BN,
                         (unsigned long long)C_DIM, PAD_W);
    }

    // ---------------- preload x A-fragments (ISA 16-bit A 16x32 layout) ----------
    v16bf xa[8];
    {
        const float* xrow = x + (size_t)(rowBlock + wave * 16 + ln) * IN_DIM;
        #pragma unroll
        for (int ks = 0; ks < 8; ++ks) {
            const int k0 = ks * 32 + hi * 8;
            float4 a0 = *(const float4*)(xrow + k0);
            float4 a1 = *(const float4*)(xrow + k0 + 4);
            float4 b0 = *(const float4*)(xrow + k0 + 16);
            float4 b1 = *(const float4*)(xrow + k0 + 20);
            v16bf f;
            f[0]  = tobf(a0.x); f[1]  = tobf(a0.y); f[2]  = tobf(a0.z); f[3]  = tobf(a0.w);
            f[4]  = tobf(a1.x); f[5]  = tobf(a1.y); f[6]  = tobf(a1.z); f[7]  = tobf(a1.w);
            f[8]  = tobf(b0.x); f[9]  = tobf(b0.y); f[10] = tobf(b0.z); f[11] = tobf(b0.w);
            f[12] = tobf(b1.x); f[13] = tobf(b1.y); f[14] = tobf(b1.z); f[15] = tobf(b1.w);
            xa[ks] = f;
        }
    }

    // D-layout: VGPR r <-> row (r + 8*hi); 8 consecutive f32 loads
    float xsqv[8];
    {
        const float* xp = xsq_g + rowBlock + wave * 16 + hi * 8;
        #pragma unroll
        for (int r = 0; r < 8; ++r) xsqv[r] = xp[r];
    }

    v8f acc[8];
    #pragma unroll
    for (int i = 0; i < 8; ++i) acc[i] = {};

    __bf16* scr = &Bscr[wave][0];

    // ================= main loop: TDM double buffer =================
    for (int ch = 0; ch < NCHUNK; ++ch) {
        const int cc  = ch * BC;
        const int buf = ch & 1;

        __syncthreads();  // everyone done reading buf^1 (chunk ch-1)

        if (wave == 0) {
            // issue chunk ch+1 into the other buffer (last iter: harmless dummy)
            const int ncc = ((ch + 1) & (NCHUNK - 1)) * BC;
            const unsigned clN = (unsigned)(uintptr_t)(void*)&Clds[buf ^ 1][0];
            const unsigned wlN = (unsigned)(uintptr_t)(void*)&Wlds[buf ^ 1][0];
            tdm_load_bf16_2d(clN, Cb + (size_t)ncc * IN_DIM, (unsigned)(C_DIM * IN_DIM), 1u,
                             (unsigned)(BC * IN_DIM), 1u,
                             (unsigned long long)(BC * IN_DIM), PAD_C);
            tdm_load_bf16_2d(wlN, Wt + (size_t)outBase * C_DIM + ncc, (unsigned)C_DIM,
                             (unsigned)OUT_DIM, (unsigned)BC, (unsigned)BN,
                             (unsigned long long)C_DIM, PAD_W);
            // TDM completes in order: <=2 outstanding means chunk ch landed
            __builtin_amdgcn_s_wait_tensorcnt(2);
        }

        const float csq0 = csq_g[cc + ln];
        const float csq1 = csq_g[cc + 16 + ln];

        __syncthreads();  // chunk ch data visible to all waves

        // -------- stage 1: S = x . c^T  (two 16x16 N-tiles, K=256) --------
        v8f S0 = {}, S1 = {};
        #pragma unroll
        for (int ks = 0; ks < 8; ++ks) {
            v16bf b0 = ld_frag(&Clds[buf][(ln)      * CL_STRIDE + ks * 32 + hi * 8]);
            v16bf b1 = ld_frag(&Clds[buf][(16 + ln) * CL_STRIDE + ks * 32 + hi * 8]);
            S0 = __builtin_amdgcn_wmma_f32_16x16x32_bf16(false, xa[ks], false, b0,
                                                         (short)0, S0, false, false);
            S1 = __builtin_amdgcn_wmma_f32_16x16x32_bf16(false, xa[ks], false, b1,
                                                         (short)0, S1, false, false);
        }

        // -------- RBF: exp(-max(d2,0)) = exp(min(2S - |x|^2 - |c|^2, 0)) --------
        #pragma unroll
        for (int r = 0; r < 8; ++r) {
            const int row = r + 8 * hi;
            float e0 = __expf(ALPHA_F * fminf(2.f * S0[r] - xsqv[r] - csq0, 0.f));
            float e1 = __expf(ALPHA_F * fminf(2.f * S1[r] - xsqv[r] - csq1, 0.f));
            scr[row * BS_STRIDE + ln]      = tobf(e0);
            scr[row * BS_STRIDE + 16 + ln] = tobf(e1);
        }
        asm volatile("" ::: "memory");  // same-wave LDS is in-order; stop reordering

        v16bf ba;
        {
            v8bf blo = *(const v8bf*)&scr[ln * BS_STRIDE + hi * 8];
            v8bf bhi = *(const v8bf*)&scr[ln * BS_STRIDE + 16 + hi * 8];
            #pragma unroll
            for (int i = 0; i < 8; ++i) { ba[i] = blo[i]; ba[8 + i] = bhi[i]; }
        }

        // -------- stage 2: out += basis . W  (8 N-tiles, K=32) --------
        #pragma unroll
        for (int nt = 0; nt < 8; ++nt) {
            v16bf wb = ld_frag(&Wlds[buf][(nt * 16 + ln) * WL_STRIDE + hi * 8]);
            acc[nt] = __builtin_amdgcn_wmma_f32_16x16x32_bf16(false, ba, false, wb,
                                                              (short)0, acc[nt], false, false);
        }
    }

    // ---------------- writeback (D layout), non-temporal ----------------
    const int rowBase = rowBlock + wave * 16 + 8 * hi;
    #pragma unroll
    for (int nt = 0; nt < 8; ++nt) {
        #pragma unroll
        for (int r = 0; r < 8; ++r) {
            __builtin_nontemporal_store(
                acc[nt][r],
                out + (size_t)(rowBase + r) * OUT_DIM + outBase + nt * 16 + ln);
        }
    }
}

// ------------------------------------------------------------------
// Fallback (no workspace): self-contained, in-kernel staging (padded)
// ------------------------------------------------------------------
__global__ __launch_bounds__(256, 1)
void rbf_kan_fused(const float* __restrict__ x,
                   const float* __restrict__ centers,
                   const float* __restrict__ weights,
                   float* __restrict__ out)
{
    __shared__ __align__(16) __bf16 Clds[BC * CL_STRIDE];
    __shared__ __align__(16) __bf16 Wlds[BN * WL_STRIDE];
    __shared__ __align__(16) __bf16 Bscr[NWAVES][16 * BS_STRIDE];
    __shared__ float csq_s[C_DIM];
    __shared__ float xsq_s[BM];

    const int tid      = threadIdx.x;
    const int wave     = tid >> 5;
    const int lane     = tid & 31;
    const int hi       = lane >> 4;
    const int ln       = lane & 15;
    const int rowBlock = blockIdx.x * BM;
    const int outBase  = blockIdx.y * BN;

    if (tid < BM) {
        const float4* xr = (const float4*)(x + (size_t)(rowBlock + tid) * IN_DIM);
        float s = 0.f;
        #pragma unroll 8
        for (int i = 0; i < IN_DIM / 4; ++i) {
            float4 v = xr[i];
            s += v.x * v.x + v.y * v.y + v.z * v.z + v.w * v.w;
        }
        xsq_s[tid] = s;
    }
    for (int c = tid; c < C_DIM; c += 256) {
        const float4* cr = (const float4*)(centers + (size_t)c * IN_DIM);
        float s = 0.f;
        #pragma unroll 8
        for (int i = 0; i < IN_DIM / 4; ++i) {
            float4 v = cr[i];
            s += v.x * v.x + v.y * v.y + v.z * v.z + v.w * v.w;
        }
        csq_s[c] = s;
    }

    v16bf xa[8];
    {
        const float* xrow = x + (size_t)(rowBlock + wave * 16 + ln) * IN_DIM;
        #pragma unroll
        for (int ks = 0; ks < 8; ++ks) {
            const int k0 = ks * 32 + hi * 8;
            float4 a0 = *(const float4*)(xrow + k0);
            float4 a1 = *(const float4*)(xrow + k0 + 4);
            float4 b0 = *(const float4*)(xrow + k0 + 16);
            float4 b1 = *(const float4*)(xrow + k0 + 20);
            v16bf f;
            f[0]  = tobf(a0.x); f[1]  = tobf(a0.y); f[2]  = tobf(a0.z); f[3]  = tobf(a0.w);
            f[4]  = tobf(a1.x); f[5]  = tobf(a1.y); f[6]  = tobf(a1.z); f[7]  = tobf(a1.w);
            f[8]  = tobf(b0.x); f[9]  = tobf(b0.y); f[10] = tobf(b0.z); f[11] = tobf(b0.w);
            f[12] = tobf(b1.x); f[13] = tobf(b1.y); f[14] = tobf(b1.z); f[15] = tobf(b1.w);
            xa[ks] = f;
        }
    }

    v8f acc[8];
    #pragma unroll
    for (int i = 0; i < 8; ++i) acc[i] = {};

    __syncthreads();

    float xsqv[8];
    #pragma unroll
    for (int r = 0; r < 8; ++r) xsqv[r] = xsq_s[wave * 16 + hi * 8 + r];

    __bf16* scr = &Bscr[wave][0];

    for (int cc = 0; cc < C_DIM; cc += BC) {
        __syncthreads();

        #pragma unroll 4
        for (int i = 0; i < BC; ++i)
            Clds[i * CL_STRIDE + tid] = tobf(centers[(size_t)(cc + i) * IN_DIM + tid]);

        #pragma unroll 4
        for (int i = 0; i < 16; ++i) {
            const int idx = tid + i * 256;
            const int k   = idx >> 7;
            const int j   = idx & 127;
            Wlds[j * WL_STRIDE + k] = tobf(weights[(size_t)(cc + k) * OUT_DIM + outBase + j]);
        }

        const float csq0 = csq_s[cc + ln];
        const float csq1 = csq_s[cc + 16 + ln];

        __syncthreads();

        v8f S0 = {}, S1 = {};
        #pragma unroll
        for (int ks = 0; ks < 8; ++ks) {
            v16bf b0 = ld_frag(&Clds[(ln)      * CL_STRIDE + ks * 32 + hi * 8]);
            v16bf b1 = ld_frag(&Clds[(16 + ln) * CL_STRIDE + ks * 32 + hi * 8]);
            S0 = __builtin_amdgcn_wmma_f32_16x16x32_bf16(false, xa[ks], false, b0,
                                                         (short)0, S0, false, false);
            S1 = __builtin_amdgcn_wmma_f32_16x16x32_bf16(false, xa[ks], false, b1,
                                                         (short)0, S1, false, false);
        }

        #pragma unroll
        for (int r = 0; r < 8; ++r) {
            const int row = r + 8 * hi;
            float e0 = __expf(ALPHA_F * fminf(2.f * S0[r] - xsqv[r] - csq0, 0.f));
            float e1 = __expf(ALPHA_F * fminf(2.f * S1[r] - xsqv[r] - csq1, 0.f));
            scr[row * BS_STRIDE + ln]      = tobf(e0);
            scr[row * BS_STRIDE + 16 + ln] = tobf(e1);
        }
        asm volatile("" ::: "memory");

        v16bf ba;
        {
            v8bf blo = *(const v8bf*)&scr[ln * BS_STRIDE + hi * 8];
            v8bf bhi = *(const v8bf*)&scr[ln * BS_STRIDE + 16 + hi * 8];
            #pragma unroll
            for (int i = 0; i < 8; ++i) { ba[i] = blo[i]; ba[8 + i] = bhi[i]; }
        }

        #pragma unroll
        for (int nt = 0; nt < 8; ++nt) {
            v16bf wb = ld_frag(&Wlds[(nt * 16 + ln) * WL_STRIDE + hi * 8]);
            acc[nt] = __builtin_amdgcn_wmma_f32_16x16x32_bf16(false, ba, false, wb,
                                                              (short)0, acc[nt], false, false);
        }
    }

    const int rowBase = rowBlock + wave * 16 + 8 * hi;
    #pragma unroll
    for (int nt = 0; nt < 8; ++nt) {
        #pragma unroll
        for (int r = 0; r < 8; ++r) {
            __builtin_nontemporal_store(
                acc[nt][r],
                out + (size_t)(rowBase + r) * OUT_DIM + outBase + nt * 16 + ln);
        }
    }
}

extern "C" void kernel_launch(void* const* d_in, const int* in_sizes, int n_in,
                              void* d_out, int out_size, void* d_ws, size_t ws_size,
                              hipStream_t stream) {
    const float* x       = (const float*)d_in[0];
    const float* centers = (const float*)d_in[1];
    const float* weights = (const float*)d_in[2];
    float*       out     = (float*)d_out;

    const int n = in_sizes[0] / IN_DIM;          // 16384
    dim3 grid(n / BM, OUT_DIM / BN);             // (128, 4)
    dim3 block(256);                             // 8 wave32

    const size_t cbBytes  = (size_t)C_DIM * IN_DIM * sizeof(__bf16);   // 512 KB
    const size_t wtBytes  = (size_t)OUT_DIM * C_DIM * sizeof(__bf16);  //   1 MB
    const size_t csqBytes = (size_t)C_DIM * sizeof(float);             //   4 KB
    const size_t xsqBytes = (size_t)n * sizeof(float);                 //  64 KB

    if (ws_size >= cbBytes + wtBytes + csqBytes + xsqBytes) {
        char* ws = (char*)d_ws;
        __bf16* Cb  = (__bf16*)ws;
        __bf16* Wt  = (__bf16*)(ws + cbBytes);
        float*  csq = (float*)(ws + cbBytes + wtBytes);
        float*  xsq = (float*)(ws + cbBytes + wtBytes + csqBytes);

        const int prepThreads = C_DIM * IN_DIM + OUT_DIM * C_DIM;      // 786432
        prep_bf16<<<prepThreads / 256, 256, 0, stream>>>(centers, weights, Cb, Wt);
        prep_norms<<<(n + C_DIM) / 256, 256, 0, stream>>>(x, centers, xsq, csq, n);
        rbf_kan_fused_tdm<<<grid, block, 0, stream>>>(x, Cb, Wt, csq, xsq, out);
    } else {
        rbf_kan_fused<<<grid, block, 0, stream>>>(x, centers, weights, out);
    }
}
